// Connector_77738908057780
// MI455X (gfx1250) — compile-verified, matrix operand order checked by reading, not run on hardware
//
#include <hip/hip_runtime.h>
#include <hip/hip_bf16.h>
#include <math.h>

// ---------------------------------------------------------------------------
// Problem constants (SigLIP connector): H=1152, B=8, N=729, L=26
// ---------------------------------------------------------------------------
#define HDIM   1152
#define MROWS  5832          // B*N = 8*729
#define K1DIM  3456          // 3*H
#define NLAYER 13            // layers per segment mean

typedef __attribute__((ext_vector_type(16))) __bf16   v16bf;
typedef __attribute__((ext_vector_type(8)))  float    v8f;
typedef __attribute__((ext_vector_type(4)))  unsigned u32x4;

union Frag16 { v16bf v; u32x4 u[2]; };
union BF8    { u32x4 u; __hip_bfloat16 h[8]; };

__device__ __constant__ float NF4_LUT[16] = {
    -1.0f, -0.6961928009986877f, -0.5250730514526367f, -0.39491748809814453f,
    -0.28444138169288635f, -0.18477343022823334f, -0.09105003625154495f, 0.0f,
    0.07958029955625534f, 0.16093020141124725f, 0.24611230194568634f,
    0.33791524171829224f, 0.4407098591327667f, 0.5626170039176941f,
    0.7229568362236023f, 1.0f };

// ---------------------------------------------------------------------------
// Kernel 1: build cat = [x | mean(hidden[0:13]) | mean(hidden[13:26])]
// ---------------------------------------------------------------------------
__global__ __launch_bounds__(256) void build_cat_kernel(
    const __hip_bfloat16* __restrict__ x,
    const __hip_bfloat16* __restrict__ hidden,
    __hip_bfloat16* __restrict__ cat)
{
    const long long chunk  = (long long)blockIdx.x * blockDim.x + threadIdx.x;
    const long long nChunk = (long long)MROWS * HDIM / 8;
    if (chunk >= nChunk) return;
    const long long e    = chunk * 8;
    const long long m    = e / HDIM;
    const int       hcol = (int)(e % HDIM);

    BF8 vx = *(const BF8*)(x + e);
    *(BF8*)(cat + m * K1DIM + hcol) = vx;

    const float inv13 = 1.0f / 13.0f;
    float a1[8] = {0,0,0,0,0,0,0,0};
    float a2[8] = {0,0,0,0,0,0,0,0};
    #pragma unroll 1
    for (int l = 0; l < NLAYER; ++l) {
        BF8 v = *(const BF8*)(hidden + ((long long)l * MROWS + m) * HDIM + hcol);
        #pragma unroll
        for (int i = 0; i < 8; ++i) a1[i] += __bfloat162float(v.h[i]);
    }
    #pragma unroll 1
    for (int l = NLAYER; l < 2 * NLAYER; ++l) {
        BF8 v = *(const BF8*)(hidden + ((long long)l * MROWS + m) * HDIM + hcol);
        #pragma unroll
        for (int i = 0; i < 8; ++i) a2[i] += __bfloat162float(v.h[i]);
    }
    BF8 o1, o2;
    #pragma unroll
    for (int i = 0; i < 8; ++i) {
        o1.h[i] = __float2bfloat16(a1[i] * inv13);
        o2.h[i] = __float2bfloat16(a2[i] * inv13);
    }
    *(BF8*)(cat + m * K1DIM + HDIM     + hcol) = o1;
    *(BF8*)(cat + m * K1DIM + 2 * HDIM + hcol) = o2;
}

// ---------------------------------------------------------------------------
// Kernel 2: NF4 dequant  w[o,k] = LUT[code[o,k]] * scales[o, k/64]  -> bf16
// ---------------------------------------------------------------------------
__global__ __launch_bounds__(256) void dequant_nf4_kernel(
    const int*   __restrict__ codes,
    const float* __restrict__ scales,
    __hip_bfloat16* __restrict__ w,
    int total, int K)
{
    const int base = (blockIdx.x * blockDim.x + threadIdx.x) * 8;
    if (base >= total) return;
    const int row = base / K;
    const int col = base % K;
    const float s = scales[row * (K / 64) + (col >> 6)];
    BF8 o;
    #pragma unroll
    for (int j = 0; j < 8; ++j)
        o.h[j] = __float2bfloat16(NF4_LUT[codes[base + j] & 15] * s);
    *(BF8*)(w + base) = o;
}

// ---------------------------------------------------------------------------
// Kernel 3: bf16 WMMA GEMM with LDS double-buffering.
//   Out[m,n] = act( sum_k A[m,k]*W[n,k] + bias[n] )
//   Block tile 128(M) x 128(N), K-stage 64; 8 waves as 2(M) x 4(N),
//   each wave owns 64x32 -> 16 v_wmma_f32_16x16x32_bf16 per stage.
//   EPI=0: bias + exact GELU, bf16 out.   EPI=1: bias only, f32 out.
// ---------------------------------------------------------------------------
template <int EPI>
__global__ __launch_bounds__(256) void gemm_wmma_kernel(
    const __hip_bfloat16* __restrict__ A,    // [Mtot, K]
    const __hip_bfloat16* __restrict__ W,    // [Ntot, K]
    const __hip_bfloat16* __restrict__ bias, // [Ntot]
    void* __restrict__ Out, int Mtot, int Ntot, int K)
{
    constexpr int BM = 128, BN = 128, BK = 64;
    constexpr int RQ = 9;                       // u32x4 per LDS row (72 bf16 = 144B)
    __shared__ u32x4 sA[2][BM * RQ];            // 2 x 18 KB
    __shared__ u32x4 sB[2][BN * RQ];            // 2 x 18 KB

    const int tid   = threadIdx.x;
    const int wave  = tid >> 5;
    const int lane  = tid & 31;
    const int waveM = wave >> 2;                // 0..1 -> 64-row strip
    const int waveN = wave & 3;                 // 0..3 -> 32-col strip
    const int m0    = blockIdx.x * BM;
    const int n0    = blockIdx.y * BN;
    const int lrow  = lane & 15;
    const int lhalf = lane >> 4;

    // ---- stage-invariant staging coordinates: 1024 x 16B per matrix ----
    // thread t loads rows (t>>3)+32*i, segment t&7 (seg*8 bf16 within 64-k row)
    const int seg = tid & 7;
    const int r0  = tid >> 3;
    const __hip_bfloat16* pA[4];
    const __hip_bfloat16* pB[4];
    bool gOK[4];
    #pragma unroll
    for (int t = 0; t < 4; ++t) {
        const int gm = m0 + r0 + 32 * t;
        const int gn = n0 + r0 + 32 * t;
        gOK[t] = (gm < Mtot);
        pA[t]  = A + (long long)gm * K + seg * 8;
        pB[t]  = W + (long long)gn * K + seg * 8;
    }

    const u32x4 zero = {0u, 0u, 0u, 0u};
    v8f acc[4][2] = {};
    const int nst = K / BK;

    // ---- prologue: stage 0 -> LDS buffer 0 ----
    #pragma unroll
    for (int t = 0; t < 4; ++t) {
        sA[0][(r0 + 32 * t) * RQ + seg] = gOK[t] ? *(const u32x4*)(pA[t]) : zero;
        sB[0][(r0 + 32 * t) * RQ + seg] = *(const u32x4*)(pB[t]);
    }

    for (int s = 0; s < nst; ++s) {
        __syncthreads();
        const int  buf  = s & 1;
        const bool more = (s + 1 < nst);
        const int  kN   = (s + 1) * BK;

        // ---- prefetch stage s+1 from global into registers ----
        u32x4 pa[4], pb[4];
        if (more) {
            #pragma unroll
            for (int t = 0; t < 4; ++t) {
                pa[t] = gOK[t] ? *(const u32x4*)(pA[t] + kN) : zero;
                pb[t] = *(const u32x4*)(pB[t] + kN);
            }
        }

        // ---- compute on buffer `buf`: 2 k-chunks x 8 WMMA ----
        #pragma unroll
        for (int c = 0; c < 2; ++c) {
            Frag16 fa[4], fb[2];
            #pragma unroll
            for (int i = 0; i < 4; ++i) {
                const int r = waveM * 64 + i * 16 + lrow;
                fa[i].u[0] = sA[buf][r * RQ + c * 4 + lhalf];
                fa[i].u[1] = sA[buf][r * RQ + c * 4 + lhalf + 2];
            }
            #pragma unroll
            for (int j = 0; j < 2; ++j) {
                const int r = waveN * 32 + j * 16 + lrow;
                fb[j].u[0] = sB[buf][r * RQ + c * 4 + lhalf];
                fb[j].u[1] = sB[buf][r * RQ + c * 4 + lhalf + 2];
            }
            #pragma unroll
            for (int i = 0; i < 4; ++i)
                #pragma unroll
                for (int j = 0; j < 2; ++j)
                    acc[i][j] = __builtin_amdgcn_wmma_f32_16x16x32_bf16(
                        false, fa[i].v, false, fb[j].v,
                        (short)0, acc[i][j], false, false);
        }

        // ---- write prefetched stage to the other buffer ----
        if (more) {
            #pragma unroll
            for (int t = 0; t < 4; ++t) {
                sA[buf ^ 1][(r0 + 32 * t) * RQ + seg] = pa[t];
                sB[buf ^ 1][(r0 + 32 * t) * RQ + seg] = pb[t];
            }
        }
    }

    // ---- epilogue: C layout VGPR r -> M = r (lanes 0-15) / r+8 (16-31) ----
    #pragma unroll
    for (int i = 0; i < 4; ++i) {
        #pragma unroll
        for (int j = 0; j < 2; ++j) {
            #pragma unroll
            for (int r = 0; r < 8; ++r) {
                const int gm = m0 + waveM * 64 + i * 16 + lhalf * 8 + r;
                const int gn = n0 + waveN * 32 + j * 16 + lrow;
                if (gm < Mtot) {
                    float v = acc[i][j][r] + __bfloat162float(bias[gn]);
                    if (EPI == 0) {
                        v = 0.5f * v * (1.0f + erff(v * 0.7071067811865476f));
                        ((__hip_bfloat16*)Out)[(long long)gm * Ntot + gn] =
                            __float2bfloat16(v);
                    } else {
                        ((float*)Out)[(long long)gm * Ntot + gn] = v;
                    }
                }
            }
        }
    }
}

// ---------------------------------------------------------------------------
// Kernel 4: per-row dual LayerNorm + sigmoid gate
// ---------------------------------------------------------------------------
__global__ __launch_bounds__(256) void ln_gate_kernel(
    const __hip_bfloat16* __restrict__ x,   // [MROWS, H]
    const float*          __restrict__ fg,  // [MROWS, H]
    const __hip_bfloat16* __restrict__ g1,
    const __hip_bfloat16* __restrict__ b1,
    const __hip_bfloat16* __restrict__ g2,
    const __hip_bfloat16* __restrict__ b2,
    const __hip_bfloat16* __restrict__ alpha,
    __hip_bfloat16* __restrict__ out)
{
    const int m   = blockIdx.x;
    const int tid = threadIdx.x;
    const __hip_bfloat16* xr = x  + (long long)m * HDIM;
    const float*          fr = fg + (long long)m * HDIM;

    float sx = 0.f, sxx = 0.f, sf = 0.f, sff = 0.f;
    for (int i = tid; i < HDIM; i += 256) {
        const float xv = __bfloat162float(xr[i]);
        const float fv = fr[i];
        sx += xv; sxx += xv * xv;
        sf += fv; sff += fv * fv;
    }

    __shared__ float red[256];
    auto block_sum = [&](float v) -> float {
        red[tid] = v; __syncthreads();
        for (int s = 128; s > 0; s >>= 1) {
            if (tid < s) red[tid] += red[tid + s];
            __syncthreads();
        }
        const float r = red[0]; __syncthreads();
        return r;
    };
    sx  = block_sum(sx);
    sxx = block_sum(sxx);
    sf  = block_sum(sf);
    sff = block_sum(sff);

    const float invH = 1.0f / (float)HDIM;
    const float mu1  = sx * invH;
    const float mu2  = sf * invH;
    const float rs1  = rsqrtf(fmaxf(sxx * invH - mu1 * mu1, 0.f) + 1e-5f);
    const float rs2  = rsqrtf(fmaxf(sff * invH - mu2 * mu2, 0.f) + 1e-5f);

    const float af = __bfloat162float(alpha[0]);
    const float wf = __bfloat162float(
        __float2bfloat16(1.0f / (1.0f + expf(-af))));   // bf16-rounded gate

    for (int i = tid; i < HDIM; i += 256) {
        const float xv = __bfloat162float(xr[i]);
        const float fv = fr[i];
        float n1 = (xv - mu1) * rs1 * __bfloat162float(g1[i]) +
                   __bfloat162float(b1[i]);
        float n2 = (fv - mu2) * rs2 * __bfloat162float(g2[i]) +
                   __bfloat162float(b2[i]);
        n1 = __bfloat162float(__float2bfloat16(n1));
        n2 = __bfloat162float(__float2bfloat16(n2));
        out[(long long)m * HDIM + i] =
            __float2bfloat16(wf * n2 + (1.0f - wf) * n1);
    }
}

// ---------------------------------------------------------------------------
// Host launcher
// ---------------------------------------------------------------------------
extern "C" void kernel_launch(void* const* d_in, const int* in_sizes, int n_in,
                              void* d_out, int out_size, void* d_ws, size_t ws_size,
                              hipStream_t stream)
{
    const __hip_bfloat16* x      = (const __hip_bfloat16*)d_in[0];
    const __hip_bfloat16* hidden = (const __hip_bfloat16*)d_in[1];
    const int*            codes1 = (const int*)d_in[2];
    const float*          scl1   = (const float*)d_in[3];
    const __hip_bfloat16* b1     = (const __hip_bfloat16*)d_in[4];
    const int*            codes2 = (const int*)d_in[5];
    const float*          scl2   = (const float*)d_in[6];
    const __hip_bfloat16* b2     = (const __hip_bfloat16*)d_in[7];
    const __hip_bfloat16* ln1g   = (const __hip_bfloat16*)d_in[8];
    const __hip_bfloat16* ln1b   = (const __hip_bfloat16*)d_in[9];
    const __hip_bfloat16* ln2g   = (const __hip_bfloat16*)d_in[10];
    const __hip_bfloat16* ln2b   = (const __hip_bfloat16*)d_in[11];
    const __hip_bfloat16* alpha  = (const __hip_bfloat16*)d_in[12];
    __hip_bfloat16* out = (__hip_bfloat16*)d_out;

    // workspace layout (bytes)
    char* ws = (char*)d_ws;
    const size_t CAT_B  = (size_t)MROWS * K1DIM * 2;   // 40.3 MB bf16
    const size_t W1_B   = (size_t)HDIM * K1DIM * 2;    //  8.0 MB bf16
    const size_t W2_B   = (size_t)HDIM * HDIM * 2;     //  2.7 MB bf16
    const size_t H_B    = (size_t)MROWS * HDIM * 2;    // 13.4 MB bf16
    __hip_bfloat16* cat = (__hip_bfloat16*)(ws);
    __hip_bfloat16* w1d = (__hip_bfloat16*)(ws + CAT_B);
    __hip_bfloat16* w2d = (__hip_bfloat16*)(ws + CAT_B + W1_B);
    __hip_bfloat16* hbuf= (__hip_bfloat16*)(ws + CAT_B + W1_B + W2_B);
    float*          fg  = (float*)         (ws + CAT_B + W1_B + W2_B + H_B);

    // 1) streaming reduction: build cat  (~363 MB read @ 23.3 TB/s)
    {
        const long long chunks = (long long)MROWS * HDIM / 8;
        const int blocks = (int)((chunks + 255) / 256);
        build_cat_kernel<<<blocks, 256, 0, stream>>>(x, hidden, cat);
    }
    // 2) NF4 dequant of both weight matrices
    {
        const int t1 = HDIM * K1DIM;
        dequant_nf4_kernel<<<(t1 / 8 + 255) / 256, 256, 0, stream>>>(
            codes1, scl1, w1d, t1, K1DIM);
        const int t2 = HDIM * HDIM;
        dequant_nf4_kernel<<<(t2 / 8 + 255) / 256, 256, 0, stream>>>(
            codes2, scl2, w2d, t2, HDIM);
    }
    // 3) GEMM1: h = GELU(cat @ w1^T + b1)   [5832x3456]x[3456x1152]
    {
        dim3 grid((MROWS + 127) / 128, HDIM / 128);
        gemm_wmma_kernel<0><<<grid, 256, 0, stream>>>(
            cat, w1d, b1, (void*)hbuf, MROWS, HDIM, K1DIM);
    }
    // 4) GEMM2: fg = h @ w2^T + b2          [5832x1152]x[1152x1152] -> f32
    {
        dim3 grid((MROWS + 127) / 128, HDIM / 128);
        gemm_wmma_kernel<1><<<grid, 256, 0, stream>>>(
            hbuf, w2d, b2, (void*)fg, MROWS, HDIM, HDIM);
    }
    // 5) dual LayerNorm + sigmoid gate
    ln_gate_kernel<<<MROWS, 256, 0, stream>>>(
        x, fg, ln1g, ln1b, ln2g, ln2b, alpha, out);

    (void)in_sizes; (void)n_in; (void)out_size; (void)ws_size;
}
